// HIST_2499670966382
// MI455X (gfx1250) — matrix operand order repaired and checked by reading, not compiled, not run on hardware
//
#include <hip/hip_runtime.h>
#include <hip/hip_bf16.h>

// ---------------------------------------------------------------------------
// HIST model forward for MI455X (gfx1250, wave32, WMMA 16x16x32 f16->f32).
// All matmuls are NT-GEMMs (A[MxK] row-major, B[NxK] row-major) so both WMMA
// operand fragment loads are contiguous (this code's k = 16*(lane/16) + i order, same in A and B).
// GRU hidden state is kept in per-wave LDS (f16) to avoid register spills.
// ---------------------------------------------------------------------------

typedef _Float16 f16;
typedef __attribute__((ext_vector_type(16))) _Float16 v16h;
typedef __attribute__((ext_vector_type(8)))  float    v8f;

#define NN_   8192
#define TT_   60
#define DD_   6
#define HH_   64
#define CC_   128

__device__ __forceinline__ v8f wmma32(v16h a, v16h b, v8f c) {
  return __builtin_amdgcn_wmma_f32_16x16x32_f16(false, a, false, b, (short)0, c,
                                                false, false);
}

// A fragment (16x32 f16, MxK): lane<16 -> M=lane, halfs0..15=K0..15
//                              lane>=16 -> M=lane-16, halfs0..15=K16..31
__device__ __forceinline__ v16h load_a_tile(const f16* A, int ld, int m0, int k0,
                                            int lane) {
  int m  = lane & 15;
  int kb = (lane < 16) ? 0 : 16;
  const f16* p = A + (size_t)(m0 + m) * ld + k0;
  v16h r;
#pragma unroll
  for (int i = 0; i < 8; ++i) r[i] = p[kb + i];
#pragma unroll
  for (int i = 0; i < 8; ++i) r[8 + i] = p[kb + 8 + i];
  return r;
}

// B fragment (32x16 f16, KxN) for NT: element(k,n)=W[row0+n][k0+k]
// lane<16 -> n=lane, K=0..15 ; lane>=16 -> n=lane-16, K=16..31 (contiguous)
__device__ __forceinline__ v16h load_nt_tile(const f16* W, int ld, int row0,
                                             int k0, int lane) {
  int n    = lane & 15;
  int koff = (lane < 16) ? 0 : 16;
  const f16* p = W + (size_t)(row0 + n) * ld + k0 + koff;
  v16h r;
#pragma unroll
  for (int i = 0; i < 16; ++i) r[i] = p[i];
  return r;
}

__device__ __forceinline__ float fsigm(float x) {
  return __fdividef(1.f, 1.f + __expf(-x));
}
__device__ __forceinline__ float ftanh(float x) {
  float e = __expf(-2.f * x);
  return __fdividef(1.f - e, 1.f + e);
}
#define NEGINF (-__builtin_inff())

// ---------------------------------------------------------------------------
// Fused 2-layer GRU. Block = 8 waves x 16 rows = 128 sequences. Hidden state
// lives in per-wave LDS (f16); gate tiles are computed one at a time
// (unroll 1) so only 4 f32 accumulators are live -> no spills.
// ---------------------------------------------------------------------------
__global__ __launch_bounds__(256) void k_gru(
    const float* __restrict__ x,
    const f16* __restrict__ Wih0p,  // [192][32] (D=6 zero-padded)
    const f16* __restrict__ Whh0,   // [192][64]
    const float* __restrict__ bih0, const float* __restrict__ bhh0,
    const f16* __restrict__ Wih1,   // [192][64]
    const f16* __restrict__ Whh1,   // [192][64]
    const float* __restrict__ bih1, const float* __restrict__ bhh1,
    float* __restrict__ xh, f16* __restrict__ xh16, f16* __restrict__ xhT16) {
  __shared__ f16 h0buf[8][16][HH_];
  __shared__ f16 h1buf[8][16][HH_];
  int wv = threadIdx.x >> 5, lane = threadIdx.x & 31;
  int row0 = blockIdx.x * 128 + wv * 16;
  for (int i = threadIdx.x; i < 8 * 16 * HH_; i += 256) {
    ((f16*)h0buf)[i] = (f16)0.f;
    ((f16*)h1buf)[i] = (f16)0.f;
  }
  __syncthreads();

  int col = lane & 15;
  int mh  = ((lane >> 4) & 1) << 3;

#pragma unroll 1
  for (int t = 0; t < TT_; ++t) {
    // ---- layer 0: gates = [x_t | h0] x W^T ----
    v16h ax = {};
    if (lane < 16) {
      const float* xp = x + (size_t)(row0 + (lane & 15)) * (DD_ * TT_) + t;
#pragma unroll
      for (int d = 0; d < DD_; ++d) ax[d] = (f16)xp[d * TT_];
    }
    v16h ah0 = load_a_tile(&h0buf[wv][0][0], HH_, 0, 0, lane);
    v16h ah1 = load_a_tile(&h0buf[wv][0][0], HH_, 0, 32, lane);

#pragma unroll 1
    for (int j = 0; j < 4; ++j) {
      v8f cr = {}, cz = {}, ci = {}, chn = {};
      v16h b;
      b = load_nt_tile(Wih0p, 32, j * 16, 0, lane);        cr  = wmma32(ax, b, cr);
      b = load_nt_tile(Wih0p, 32, (j + 4) * 16, 0, lane);  cz  = wmma32(ax, b, cz);
      b = load_nt_tile(Wih0p, 32, (j + 8) * 16, 0, lane);  ci  = wmma32(ax, b, ci);
      b = load_nt_tile(Whh0, 64, j * 16, 0, lane);         cr  = wmma32(ah0, b, cr);
      b = load_nt_tile(Whh0, 64, j * 16, 32, lane);        cr  = wmma32(ah1, b, cr);
      b = load_nt_tile(Whh0, 64, (j + 4) * 16, 0, lane);   cz  = wmma32(ah0, b, cz);
      b = load_nt_tile(Whh0, 64, (j + 4) * 16, 32, lane);  cz  = wmma32(ah1, b, cz);
      b = load_nt_tile(Whh0, 64, (j + 8) * 16, 0, lane);   chn = wmma32(ah0, b, chn);
      b = load_nt_tile(Whh0, 64, (j + 8) * 16, 32, lane);  chn = wmma32(ah1, b, chn);
      float bir = bih0[j * 16 + col] + bhh0[j * 16 + col];
      float biz = bih0[(j + 4) * 16 + col] + bhh0[(j + 4) * 16 + col];
      float bin = bih0[(j + 8) * 16 + col];
      float bhn = bhh0[(j + 8) * 16 + col];
#pragma unroll
      for (int e = 0; e < 8; ++e) {
        float ho = (float)h0buf[wv][mh + e][j * 16 + col];
        float r = fsigm(cr[e] + bir);
        float z = fsigm(cz[e] + biz);
        float n = ftanh(ci[e] + bin + r * (chn[e] + bhn));
        h0buf[wv][mh + e][j * 16 + col] = (f16)((1.f - z) * n + z * ho);
      }
    }

    // ---- layer 1: input = new h0 (read back as A fragments) ----
    v16h ax0 = load_a_tile(&h0buf[wv][0][0], HH_, 0, 0, lane);
    v16h ax1 = load_a_tile(&h0buf[wv][0][0], HH_, 0, 32, lane);
    v16h bh0 = load_a_tile(&h1buf[wv][0][0], HH_, 0, 0, lane);
    v16h bh1 = load_a_tile(&h1buf[wv][0][0], HH_, 0, 32, lane);

#pragma unroll 1
    for (int j = 0; j < 4; ++j) {
      v8f cr = {}, cz = {}, ci = {}, chn = {};
      v16h b;
      b = load_nt_tile(Wih1, 64, j * 16, 0, lane);         cr  = wmma32(ax0, b, cr);
      b = load_nt_tile(Wih1, 64, j * 16, 32, lane);        cr  = wmma32(ax1, b, cr);
      b = load_nt_tile(Wih1, 64, (j + 4) * 16, 0, lane);   cz  = wmma32(ax0, b, cz);
      b = load_nt_tile(Wih1, 64, (j + 4) * 16, 32, lane);  cz  = wmma32(ax1, b, cz);
      b = load_nt_tile(Wih1, 64, (j + 8) * 16, 0, lane);   ci  = wmma32(ax0, b, ci);
      b = load_nt_tile(Wih1, 64, (j + 8) * 16, 32, lane);  ci  = wmma32(ax1, b, ci);
      b = load_nt_tile(Whh1, 64, j * 16, 0, lane);         cr  = wmma32(bh0, b, cr);
      b = load_nt_tile(Whh1, 64, j * 16, 32, lane);        cr  = wmma32(bh1, b, cr);
      b = load_nt_tile(Whh1, 64, (j + 4) * 16, 0, lane);   cz  = wmma32(bh0, b, cz);
      b = load_nt_tile(Whh1, 64, (j + 4) * 16, 32, lane);  cz  = wmma32(bh1, b, cz);
      b = load_nt_tile(Whh1, 64, (j + 8) * 16, 0, lane);   chn = wmma32(bh0, b, chn);
      b = load_nt_tile(Whh1, 64, (j + 8) * 16, 32, lane);  chn = wmma32(bh1, b, chn);
      float bir = bih1[j * 16 + col] + bhh1[j * 16 + col];
      float biz = bih1[(j + 4) * 16 + col] + bhh1[(j + 4) * 16 + col];
      float bin = bih1[(j + 8) * 16 + col];
      float bhn = bhh1[(j + 8) * 16 + col];
#pragma unroll
      for (int e = 0; e < 8; ++e) {
        float ho = (float)h1buf[wv][mh + e][j * 16 + col];
        float r = fsigm(cr[e] + bir);
        float z = fsigm(cz[e] + biz);
        float n = ftanh(ci[e] + bin + r * (chn[e] + bhn));
        h1buf[wv][mh + e][j * 16 + col] = (f16)((1.f - z) * n + z * ho);
      }
    }
  }

  // epilogue: xh = last hidden of layer1 (f32 + f16 + f16-transposed)
#pragma unroll
  for (int j = 0; j < 4; ++j)
#pragma unroll
    for (int e = 0; e < 8; ++e) {
      int r = row0 + mh + e, c = j * 16 + col;
      float v = (float)h1buf[wv][mh + e][c];
      xh[(size_t)r * HH_ + c]    = v;
      xh16[(size_t)r * HH_ + c]  = (f16)v;
      xhT16[(size_t)c * NN_ + r] = (f16)v;
    }
}

// ---------------------------------------------------------------------------
// Generic NT WMMA GEMM: C[MxNc] = A[MxK] * B[NcxK]^T (+bias, +leaky-relu).
// Block = 128x64 tile, 8 waves, each wave 16 rows x 64 cols (4 C tiles).
// ---------------------------------------------------------------------------
__global__ __launch_bounds__(256) void k_gemm_nt(
    const f16* __restrict__ A, int lda, const f16* __restrict__ B, int ldb,
    const float* __restrict__ bias, float* __restrict__ C,
    f16* __restrict__ C16, f16* __restrict__ C16T, int M, int Nc, int Kd,
    int act) {
  int wv = threadIdx.x >> 5, lane = threadIdx.x & 31;
  int m0 = blockIdx.x * 128 + wv * 16;
  int n0 = blockIdx.y * 64;
  v8f acc[4] = {};
  for (int k0 = 0; k0 < Kd; k0 += 32) {
    if (k0 + 64 < Kd)
      __builtin_prefetch(A + (size_t)(m0 + (lane & 15)) * lda + k0 + 64, 0, 1);
    v16h a = load_a_tile(A, lda, m0, k0, lane);
#pragma unroll
    for (int nt = 0; nt < 4; ++nt) {
      v16h b = load_nt_tile(B, ldb, n0 + nt * 16, k0, lane);
      acc[nt] = wmma32(a, b, acc[nt]);
    }
  }
  int col = lane & 15, mh = ((lane >> 4) & 1) << 3;
#pragma unroll
  for (int nt = 0; nt < 4; ++nt) {
    int c = n0 + nt * 16 + col;
    float bb = bias ? bias[c] : 0.f;
#pragma unroll
    for (int e = 0; e < 8; ++e) {
      int r = m0 + mh + e;
      float v = acc[nt][e] + bb;
      if (act) v = v > 0.f ? v : 0.01f * v;
      if (C)    C[(size_t)r * Nc + c]    = v;
      if (C16)  C16[(size_t)r * Nc + c]  = (f16)v;
      if (C16T) C16T[(size_t)c * M + r]  = (f16)v;
    }
  }
}

// Split-K variant for tall-K reductions (C[128x64] from K=8192): grid.z walks
// K chunks, results combined with global f32 atomics (C must be pre-zeroed).
__global__ __launch_bounds__(256) void k_gemm_nt_splitk(
    const f16* __restrict__ A, int lda, const f16* __restrict__ B, int ldb,
    float* __restrict__ C, int M, int Nc, int Kd, int kchunk) {
  int wv = threadIdx.x >> 5, lane = threadIdx.x & 31;
  int m0 = blockIdx.x * 128 + wv * 16;
  int n0 = blockIdx.y * 64;
  int kb = blockIdx.z * kchunk;
  int ke = kb + kchunk; if (ke > Kd) ke = Kd;
  v8f acc[4] = {};
  for (int k0 = kb; k0 < ke; k0 += 32) {
    v16h a = load_a_tile(A, lda, m0, k0, lane);
#pragma unroll
    for (int nt = 0; nt < 4; ++nt) {
      v16h b = load_nt_tile(B, ldb, n0 + nt * 16, k0, lane);
      acc[nt] = wmma32(a, b, acc[nt]);
    }
  }
  int col = lane & 15, mh = ((lane >> 4) & 1) << 3;
#pragma unroll
  for (int nt = 0; nt < 4; ++nt) {
    int c = n0 + nt * 16 + col;
#pragma unroll
    for (int e = 0; e < 8; ++e)
      atomicAdd(&C[(size_t)(m0 + mh + e) * Nc + c], acc[nt][e]);
  }
}

// ---------------------------------------------------------------------------
// Streaming cos_sim(hsi,hsi) with online row top-3 (diag zeroed). 16 rows per
// block, 128-col chunks via WMMA; candidates merged through LDS.
// ---------------------------------------------------------------------------
__device__ __forceinline__ void ins3(float v, int j, float* bv, int* bi) {
  if (v > bv[0] || (v == bv[0] && j < bi[0])) {
    bv[2] = bv[1]; bi[2] = bi[1]; bv[1] = bv[0]; bi[1] = bi[0];
    bv[0] = v; bi[0] = j;
  } else if (v > bv[1] || (v == bv[1] && j < bi[1])) {
    bv[2] = bv[1]; bi[2] = bi[1]; bv[1] = v; bi[1] = j;
  } else if (v > bv[2] || (v == bv[2] && j < bi[2])) {
    bv[2] = v; bi[2] = j;
  }
}

__global__ __launch_bounds__(256) void k_topk(const f16* __restrict__ hsi16,
                                              const float* __restrict__ nrm,
                                              float* __restrict__ topv,
                                              int* __restrict__ topi) {
  __shared__ float Sbuf[16][128];
  __shared__ float candv[16][16][3];
  __shared__ int   candi[16][16][3];
  int wv = threadIdx.x >> 5, lane = threadIdx.x & 31;
  int row0 = blockIdx.x * 16;
  int tr = threadIdx.x >> 4, tc = threadIdx.x & 15;
  float tv[3] = {NEGINF, NEGINF, NEGINF};
  int   ti[3] = {0x7fffffff, 0x7fffffff, 0x7fffffff};
  v16h a0 = load_a_tile(hsi16, HH_, row0, 0, lane);
  v16h a1 = load_a_tile(hsi16, HH_, row0, 32, lane);
  for (int jc = 0; jc < NN_; jc += 128) {
    v16h b0 = load_nt_tile(hsi16, HH_, jc + wv * 16, 0, lane);
    v16h b1 = load_nt_tile(hsi16, HH_, jc + wv * 16, 32, lane);
    v8f s = {};
    s = wmma32(a0, b0, s);
    s = wmma32(a1, b1, s);
    int col = lane & 15, mh = ((lane >> 4) & 1) << 3;
#pragma unroll
    for (int e = 0; e < 8; ++e) {
      int r = row0 + mh + e, j = jc + wv * 16 + col;
      float den = nrm[r] * nrm[j];
      float v = (den == 0.f) ? 0.f : __fdividef(s[e], den);
      if (j == r) v = 0.f;
      Sbuf[mh + e][wv * 16 + col] = v;
    }
    __syncthreads();
#pragma unroll
    for (int i = 0; i < 8; ++i)
      ins3(Sbuf[tr][tc + 16 * i], jc + tc + 16 * i, tv, ti);
    __syncthreads();
  }
#pragma unroll
  for (int s2 = 0; s2 < 3; ++s2) { candv[tr][tc][s2] = tv[s2]; candi[tr][tc][s2] = ti[s2]; }
  __syncthreads();
  if (threadIdx.x < 16) {
    int r = threadIdx.x;
    float bv[3] = {NEGINF, NEGINF, NEGINF};
    int   bi[3] = {0x7fffffff, 0x7fffffff, 0x7fffffff};
    for (int c = 0; c < 16; ++c)
      for (int s2 = 0; s2 < 3; ++s2) ins3(candv[r][c][s2], candi[r][c][s2], bv, bi);
    for (int s2 = 0; s2 < 3; ++s2) {
      topv[(row0 + r) * 3 + s2] = bv[s2];
      topi[(row0 + r) * 3 + s2] = bi[s2];
    }
  }
}

// ---------------------------------------------------------------------------
// Flash-style streaming softmax(cos_sim(hsi,hidden2)) @ hidden2. S tiles and
// P@V via WMMA, online softmax stats in LDS; N x N never materialized.
// ---------------------------------------------------------------------------
__global__ __launch_bounds__(256) void k_flash(
    const f16* __restrict__ q16, const float* __restrict__ qn,
    const f16* __restrict__ kv16, const f16* __restrict__ kvT16,
    const float* __restrict__ kn, const float* __restrict__ valid2,
    float* __restrict__ outf, f16* __restrict__ out16) {
  __shared__ float Sbuf[16][128];
  __shared__ f16   Pbuf[16][128];
  __shared__ float mrow[16], lrow[16], srow[16];
  int wv = threadIdx.x >> 5, lane = threadIdx.x & 31;
  int row0 = blockIdx.x * 16;
  if (threadIdx.x < 16) {
    mrow[threadIdx.x] = NEGINF;
    lrow[threadIdx.x] = 0.f;
    srow[threadIdx.x] = 1.f;
  }
  v16h a0 = load_a_tile(q16, HH_, row0, 0, lane);
  v16h a1 = load_a_tile(q16, HH_, row0, 32, lane);
  v8f oacc = {};
  int col = lane & 15, mh = ((lane >> 4) & 1) << 3;
  __syncthreads();
  for (int jc = 0; jc < NN_; jc += 128) {
    v16h b0 = load_nt_tile(kv16, HH_, jc + wv * 16, 0, lane);
    v16h b1 = load_nt_tile(kv16, HH_, jc + wv * 16, 32, lane);
    v8f s = {};
    s = wmma32(a0, b0, s);
    s = wmma32(a1, b1, s);
#pragma unroll
    for (int e = 0; e < 8; ++e) {
      int r = row0 + mh + e, j = jc + wv * 16 + col;
      float den = qn[r] * kn[j];
      float v = (valid2[j] != 0.f)
                    ? ((den == 0.f) ? 0.f : __fdividef(s[e], den))
                    : NEGINF;
      Sbuf[mh + e][wv * 16 + col] = v;
    }
    __syncthreads();
    if (threadIdx.x < 16) {  // per-row online-softmax stats + P
      int r = threadIdx.x;
      float m = mrow[r], mc = m;
      for (int j = 0; j < 128; ++j) mc = fmaxf(mc, Sbuf[r][j]);
      float sc = (mc == NEGINF) ? 1.f : __expf(m - mc);
      if (m == NEGINF) sc = (mc == NEGINF) ? 1.f : 0.f;
      float l = lrow[r] * sc, ls = 0.f;
      for (int j = 0; j < 128; ++j) {
        float sv = Sbuf[r][j];
        float p = (sv == NEGINF) ? 0.f : __expf(sv - mc);
        Pbuf[r][j] = (f16)p;
        ls += p;
      }
      mrow[r] = mc;
      lrow[r] = l + ls;
      srow[r] = sc;
    }
    __syncthreads();
    if (wv < 4) {  // oacc (16 x 16 tile of the 16x64 output) update via WMMA
#pragma unroll
      for (int e = 0; e < 8; ++e) oacc[e] *= srow[mh + e];
#pragma unroll
      for (int kb = 0; kb < 4; ++kb) {
        v16h ap = load_a_tile(&Pbuf[0][0], 128, 0, kb * 32, lane);
        v16h bv = load_nt_tile(kvT16, NN_, wv * 16, jc + kb * 32, lane);
        oacc = wmma32(ap, bv, oacc);
      }
    }
    __syncthreads();
  }
  if (wv < 4) {
#pragma unroll
    for (int e = 0; e < 8; ++e) {
      int r = row0 + mh + e, c = wv * 16 + col;
      float l = lrow[mh + e];
      float v = (l > 0.f) ? __fdividef(oacc[e], l) : 0.f;
      if (outf) outf[(size_t)r * HH_ + c] = v;
      out16[(size_t)r * HH_ + c] = (f16)v;
    }
  }
}

// ------------------------- small elementwise kernels -----------------------
__global__ void k_cvt(const float* a, f16* b, int n) {
  int i = blockIdx.x * blockDim.x + threadIdx.x;
  if (i < n) b[i] = (f16)a[i];
}
__global__ void k_cvt_pad(const float* a, f16* b, int rows, int ic, int oc) {
  int i = blockIdx.x * blockDim.x + threadIdx.x;
  if (i >= rows * oc) return;
  int r = i / oc, c = i % oc;
  b[i] = (f16)((c < ic) ? a[r * ic + c] : 0.f);
}
__global__ void k_zero(float* p, int n) {
  int i = blockIdx.x * blockDim.x + threadIdx.x;
  if (i < n) p[i] = 0.f;
}
__global__ void k_rownorm(const float* a, float* nrm, int n, int h) {
  int i = blockIdx.x * blockDim.x + threadIdx.x;
  if (i >= n) return;
  float s = 0.f;
  for (int k = 0; k < h; ++k) { float v = a[(size_t)i * h + k]; s += v * v; }
  nrm[i] = sqrtf(s);
}
__global__ void k_s2c_colsum(const float* cm, const float* mv, float* cs) {
  __shared__ float red[256];
  int c = blockIdx.x;
  float s = 0.f;
  for (int n = threadIdx.x; n < NN_; n += 256) s += cm[(size_t)n * CC_ + c] * mv[n];
  red[threadIdx.x] = s; __syncthreads();
  for (int st = 128; st > 0; st >>= 1) {
    if (threadIdx.x < st) red[threadIdx.x] += red[threadIdx.x + st];
    __syncthreads();
  }
  if (threadIdx.x == 0) cs[c] = red[0];
}
__global__ void k_s2cT(const float* cm, const float* mv, const float* cs,
                       f16* s2cT) {
  int i = blockIdx.x * blockDim.x + threadIdx.x;
  if (i >= NN_ * CC_) return;
  int n = i / CC_, c = i % CC_;
  float v = cm[i] * mv[n];
  float d = cs[c] * cm[i] + 1.f;
  s2cT[(size_t)c * NN_ + n] = (f16)__fdividef(v, d);
}
__global__ void k_hidfix(float* hid, float* valid1, float* hnorm, f16* hid16,
                         f16* hidT16, int useStored) {
  __shared__ float red[64];
  int c = blockIdx.x, h = threadIdx.x;
  float v = hid[c * HH_ + h];
  red[h] = v; __syncthreads();
  for (int s = 32; s > 0; s >>= 1) {
    if (h < s) red[h] += red[h + s];
    __syncthreads();
  }
  float sum = red[0]; __syncthreads();
  float val;
  if (useStored) val = (valid1[c] != 0.f) ? v : 0.f;
  else {
    int ok = (sum != 0.f);
    if (h == 0) valid1[c] = ok ? 1.f : 0.f;
    val = ok ? v : 0.f;
  }
  hid[c * HH_ + h] = val;
  hid16[c * HH_ + h] = (f16)val;
  hidT16[(size_t)h * CC_ + c] = (f16)val;
  red[h] = val * val; __syncthreads();
  for (int s = 32; s > 0; s >>= 1) {
    if (h < s) red[h] += red[h + s];
    __syncthreads();
  }
  if (h == 0) hnorm[c] = sqrtf(red[0]);
}
__global__ void k_colsoftmax(const float* L, f16* outT) {
  __shared__ float red[256];
  int c = blockIdx.x;
  float m = NEGINF;
  for (int n = threadIdx.x; n < NN_; n += 256) m = fmaxf(m, L[(size_t)n * CC_ + c]);
  red[threadIdx.x] = m; __syncthreads();
  for (int s = 128; s > 0; s >>= 1) {
    if (threadIdx.x < s) red[threadIdx.x] = fmaxf(red[threadIdx.x], red[threadIdx.x + s]);
    __syncthreads();
  }
  m = red[0]; __syncthreads();
  float sum = 0.f;
  for (int n = threadIdx.x; n < NN_; n += 256) sum += __expf(L[(size_t)n * CC_ + c] - m);
  red[threadIdx.x] = sum; __syncthreads();
  for (int s = 128; s > 0; s >>= 1) {
    if (threadIdx.x < s) red[threadIdx.x] += red[threadIdx.x + s];
    __syncthreads();
  }
  float inv = __fdividef(1.f, red[0]);
  for (int n = threadIdx.x; n < NN_; n += 256)
    outT[(size_t)c * NN_ + n] = (f16)(__expf(L[(size_t)n * CC_ + c] - m) * inv);
}
__global__ void k_rowsoftmax(const float* XY, const float* xn, const float* hn,
                             const float* valid1, f16* probs) {
  int n = blockIdx.x * blockDim.x + threadIdx.x;
  if (n >= NN_) return;
  const float* row = XY + (size_t)n * CC_;
  float xv = xn[n];
  float m = NEGINF;
  for (int c = 0; c < CC_; ++c) {
    float den = xv * hn[c];
    float v = (valid1[c] != 0.f) ? ((den == 0.f) ? 0.f : __fdividef(row[c], den))
                                 : NEGINF;
    m = fmaxf(m, v);
  }
  float s = 0.f;
  for (int c = 0; c < CC_; ++c) {
    float den = xv * hn[c];
    float v = (valid1[c] != 0.f) ? ((den == 0.f) ? 0.f : __fdividef(row[c], den))
                                 : NEGINF;
    s += (v == NEGINF) ? 0.f : __expf(v - m);
  }
  float inv = __fdividef(1.f, s);
  for (int c = 0; c < CC_; ++c) {
    float den = xv * hn[c];
    float v = (valid1[c] != 0.f) ? ((den == 0.f) ? 0.f : __fdividef(row[c], den))
                                 : NEGINF;
    probs[(size_t)n * CC_ + c] = (f16)((v == NEGINF) ? 0.f : __expf(v - m) * inv);
  }
}
__global__ void k_hsi(const float* xh, const float* pb, float* hsi, f16* hsi16,
                      float* nrm) {
  int n = blockIdx.x * blockDim.x + threadIdx.x;
  if (n >= NN_) return;
  float s = 0.f;
  for (int h = 0; h < HH_; ++h) {
    float v = xh[(size_t)n * HH_ + h] - pb[(size_t)n * HH_ + h];
    hsi[(size_t)n * HH_ + h] = v;
    hsi16[(size_t)n * HH_ + h] = (f16)v;
    s += v * v;
  }
  nrm[n] = sqrtf(s);
}
__global__ void k_scatter(const float* tv, const int* ti, const float* hsi,
                          float* colsum, float* hid2) {
  int i = blockIdx.x * blockDim.x + threadIdx.x;
  if (i >= NN_) return;
  for (int k = 0; k < 3; ++k) {
    float v = tv[i * 3 + k];
    int j = ti[i * 3 + k];
    atomicAdd(&colsum[j], v);
    for (int h = 0; h < HH_; ++h)
      atomicAdd(&hid2[(size_t)j * HH_ + h], v * hsi[(size_t)i * HH_ + h]);
  }
}
__global__ void k_h2fix(float* hid2, const float* colsum, const float* hsin,
                        const float* hsi, f16* h216, f16* h2T16, float* h2n,
                        float* valid2) {
  int j = blockIdx.x * blockDim.x + threadIdx.x;
  if (j >= NN_) return;
  float add = (colsum[j] != 0.f && hsin[j] > 0.f) ? 1.f : 0.f;
  float vals[HH_];
  float sum = 0.f;
  for (int h = 0; h < HH_; ++h) {
    float v = hid2[(size_t)j * HH_ + h] + add * hsi[(size_t)j * HH_ + h];
    vals[h] = v; sum += v;
  }
  int ok = (sum != 0.f);
  valid2[j] = ok ? 1.f : 0.f;
  float s2 = 0.f;
  for (int h = 0; h < HH_; ++h) {
    float v = ok ? vals[h] : 0.f;
    hid2[(size_t)j * HH_ + h] = v;
    h216[(size_t)j * HH_ + h] = (f16)v;
    h2T16[(size_t)h * NN_ + j] = (f16)v;
    s2 += v * v;
  }
  h2n[j] = sqrtf(s2);
}
__global__ void k_indi(const float* xh, const float* pb, const float* hb,
                       f16* indi16, int n) {
  int i = blockIdx.x * blockDim.x + threadIdx.x;
  if (i < n) indi16[i] = (f16)(xh[i] - pb[i] - hb[i]);
}
__global__ void k_final(const float* ops, const float* ohs, const float* oin,
                        const float* w, const float* b, float* out) {
  int n = blockIdx.x * blockDim.x + threadIdx.x;
  if (n >= NN_) return;
  float s = 0.f;
  for (int h = 0; h < HH_; ++h)
    s += (ops[(size_t)n * HH_ + h] + ohs[(size_t)n * HH_ + h] +
          oin[(size_t)n * HH_ + h]) * w[h];
  out[n] = s + b[0];
}

// ---------------------------------------------------------------------------
extern "C" void kernel_launch(void* const* d_in, const int* in_sizes, int n_in,
                              void* d_out, int out_size, void* d_ws,
                              size_t ws_size, hipStream_t stream) {
  const float* x    = (const float*)d_in[0];
  const float* cm   = (const float*)d_in[1];
  const float* mv   = (const float*)d_in[2];
  const float* Wih0 = (const float*)d_in[3];
  const float* Whh0 = (const float*)d_in[4];
  const float* bih0 = (const float*)d_in[5];
  const float* bhh0 = (const float*)d_in[6];
  const float* Wih1 = (const float*)d_in[7];
  const float* Whh1 = (const float*)d_in[8];
  const float* bih1 = (const float*)d_in[9];
  const float* bhh1 = (const float*)d_in[10];
  const float* ps_w = (const float*)d_in[11];      const float* ps_b = (const float*)d_in[12];
  const float* hs_w = (const float*)d_in[13];      const float* hs_b = (const float*)d_in[14];
  const float* ps_fore_w = (const float*)d_in[15]; const float* ps_fore_b = (const float*)d_in[16];
  const float* hs_fore_w = (const float*)d_in[17]; const float* hs_fore_b = (const float*)d_in[18];
  const float* ps_back_w = (const float*)d_in[19]; const float* ps_back_b = (const float*)d_in[20];
  const float* hs_back_w = (const float*)d_in[21]; const float* hs_back_b = (const float*)d_in[22];
  const float* indi_w = (const float*)d_in[23];    const float* indi_b = (const float*)d_in[24];
  const float* out_w = (const float*)d_in[31];     const float* out_b = (const float*)d_in[32];

  char* wp = (char*)d_ws;
  auto alloc = [&](size_t bytes) -> void* {
    void* p = (void*)wp;
    wp += (bytes + 255) & ~(size_t)255;
    return p;
  };
  f16* wih0p = (f16*)alloc(192 * 32 * 2);
  f16* whh0h = (f16*)alloc(192 * 64 * 2);
  f16* wih1h = (f16*)alloc(192 * 64 * 2);
  f16* whh1h = (f16*)alloc(192 * 64 * 2);
  f16* psw16 = (f16*)alloc(64 * 64 * 2);
  f16* hsw16 = (f16*)alloc(64 * 64 * 2);
  f16* psfw16 = (f16*)alloc(64 * 64 * 2);
  f16* hsfw16 = (f16*)alloc(64 * 64 * 2);
  f16* psbw16 = (f16*)alloc(64 * 64 * 2);
  f16* hsbw16 = (f16*)alloc(64 * 64 * 2);
  f16* indiw16 = (f16*)alloc(64 * 64 * 2);
  float* xh   = (float*)alloc((size_t)NN_ * HH_ * 4);
  f16*   xh16 = (f16*)alloc((size_t)NN_ * HH_ * 2);
  f16*   xhT16 = (f16*)alloc((size_t)NN_ * HH_ * 2);
  float* xnorm = (float*)alloc(NN_ * 4);
  float* cs = (float*)alloc(CC_ * 4);
  f16* s2cT = (f16*)alloc((size_t)NN_ * CC_ * 2);
  float* hidf = (float*)alloc(CC_ * HH_ * 4);
  f16* hid16 = (f16*)alloc(CC_ * HH_ * 2);
  f16* hidT16 = (f16*)alloc(CC_ * HH_ * 2);
  float* hnorm = (float*)alloc(CC_ * 4);
  float* valid1 = (float*)alloc(CC_ * 4);
  float* Lbuf = (float*)alloc((size_t)NN_ * CC_ * 4);
  f16* s2c2T = (f16*)alloc((size_t)NN_ * CC_ * 2);
  f16* probs16 = (f16*)alloc((size_t)NN_ * CC_ * 2);
  f16* pre16 = (f16*)alloc((size_t)NN_ * HH_ * 2);
  float* pshf = (float*)alloc((size_t)NN_ * HH_ * 4);
  f16* psh16 = (f16*)alloc((size_t)NN_ * HH_ * 2);
  float* pback = (float*)alloc((size_t)NN_ * HH_ * 4);
  float* outps = (float*)alloc((size_t)NN_ * HH_ * 4);
  float* hsif = (float*)alloc((size_t)NN_ * HH_ * 4);
  f16* hsi16 = (f16*)alloc((size_t)NN_ * HH_ * 2);
  float* hsinorm = (float*)alloc(NN_ * 4);
  float* colsum = (float*)alloc(NN_ * 4);
  float* top3v = (float*)alloc((size_t)NN_ * 3 * 4);
  int*   top3i = (int*)alloc((size_t)NN_ * 3 * 4);
  float* hid2f = (float*)alloc((size_t)NN_ * HH_ * 4);
  f16* h216 = (f16*)alloc((size_t)NN_ * HH_ * 2);
  f16* h2T16 = (f16*)alloc((size_t)NN_ * HH_ * 2);
  float* h2n = (float*)alloc(NN_ * 4);
  float* valid2 = (float*)alloc(NN_ * 4);
  f16* hspre16 = (f16*)alloc((size_t)NN_ * HH_ * 2);
  float* hshf = (float*)alloc((size_t)NN_ * HH_ * 4);
  f16* hsh16 = (f16*)alloc((size_t)NN_ * HH_ * 2);
  float* hback = (float*)alloc((size_t)NN_ * HH_ * 4);
  float* ouths = (float*)alloc((size_t)NN_ * HH_ * 4);
  f16* indi16 = (f16*)alloc((size_t)NN_ * HH_ * 2);
  float* outin = (float*)alloc((size_t)NN_ * HH_ * 4);

  auto cv = [&](const float* s, f16* d, int n) {
    k_cvt<<<(n + 255) / 256, 256, 0, stream>>>(s, d, n);
  };
  k_cvt_pad<<<(192 * 32 + 255) / 256, 256, 0, stream>>>(Wih0, wih0p, 192, 6, 32);
  cv(Whh0, whh0h, 192 * 64); cv(Wih1, wih1h, 192 * 64); cv(Whh1, whh1h, 192 * 64);
  cv(ps_w, psw16, 64 * 64); cv(hs_w, hsw16, 64 * 64);
  cv(ps_fore_w, psfw16, 64 * 64); cv(hs_fore_w, hsfw16, 64 * 64);
  cv(ps_back_w, psbw16, 64 * 64); cv(hs_back_w, hsbw16, 64 * 64);
  cv(indi_w, indiw16, 64 * 64);

  // ---- GRU (fused 2 layers, WMMA) ----
  k_gru<<<NN_ / 128, 256, 0, stream>>>(x, wih0p, whh0h, bih0, bhh0, wih1h,
                                       whh1h, bih1, bhh1, xh, xh16, xhT16);
  k_rownorm<<<NN_ / 256, 256, 0, stream>>>(xh, xnorm, NN_, HH_);

  // ---- concept aggregation ----
  k_s2c_colsum<<<CC_, 256, 0, stream>>>(cm, mv, cs);
  k_s2cT<<<(NN_ * CC_ + 255) / 256, 256, 0, stream>>>(cm, mv, cs, s2cT);
  k_zero<<<(CC_ * HH_ + 255) / 256, 256, 0, stream>>>(hidf, CC_ * HH_);
  k_gemm_nt_splitk<<<dim3(1, 1, 16), 256, 0, stream>>>(s2cT, NN_, xhT16, NN_,
                                                       hidf, CC_, HH_, NN_, NN_ / 16);
  k_hidfix<<<CC_, 64, 0, stream>>>(hidf, valid1, hnorm, hid16, hidT16, 0);
  k_gemm_nt<<<dim3(NN_ / 128, 2), 256, 0, stream>>>(xh16, HH_, hid16, HH_,
                                                    nullptr, Lbuf, nullptr,
                                                    nullptr, NN_, CC_, HH_, 0);
  k_colsoftmax<<<CC_, 256, 0, stream>>>(Lbuf, s2c2T);
  k_zero<<<(CC_ * HH_ + 255) / 256, 256, 0, stream>>>(hidf, CC_ * HH_);
  k_gemm_nt_splitk<<<dim3(1, 1, 16), 256, 0, stream>>>(s2c2T, NN_, xhT16, NN_,
                                                       hidf, CC_, HH_, NN_, NN_ / 16);
  k_hidfix<<<CC_, 64, 0, stream>>>(hidf, valid1, hnorm, hid16, hidT16, 1);
  k_gemm_nt<<<dim3(NN_ / 128, 2), 256, 0, stream>>>(xh16, HH_, hid16, HH_,
                                                    nullptr, Lbuf, nullptr,
                                                    nullptr, NN_, CC_, HH_, 0);
  k_rowsoftmax<<<NN_ / 256, 256, 0, stream>>>(Lbuf, xnorm, hnorm, valid1, probs16);
  k_gemm_nt<<<dim3(NN_ / 128, 1), 256, 0, stream>>>(probs16, CC_, hidT16, CC_,
                                                    nullptr, nullptr, pre16,
                                                    nullptr, NN_, HH_, CC_, 0);
  // ---- ps chain ----
  k_gemm_nt<<<dim3(NN_ / 128, 1), 256, 0, stream>>>(pre16, HH_, psw16, HH_, ps_b,
                                                    pshf, psh16, nullptr, NN_, HH_, HH_, 0);
  k_gemm_nt<<<dim3(NN_ / 128, 1), 256, 0, stream>>>(psh16, HH_, psbw16, HH_,
                                                    ps_back_b, pback, nullptr,
                                                    nullptr, NN_, HH_, HH_, 0);
  k_gemm_nt<<<dim3(NN_ / 128, 1), 256, 0, stream>>>(psh16, HH_, psfw16, HH_,
                                                    ps_fore_b, outps, nullptr,
                                                    nullptr, NN_, HH_, HH_, 1);
  // ---- hsi / top-k / sparse aggregation ----
  k_hsi<<<NN_ / 256, 256, 0, stream>>>(xh, pback, hsif, hsi16, hsinorm);
  k_zero<<<(NN_ + 255) / 256, 256, 0, stream>>>(colsum, NN_);
  k_zero<<<(NN_ * HH_ + 255) / 256, 256, 0, stream>>>(hid2f, NN_ * HH_);
  k_topk<<<NN_ / 16, 256, 0, stream>>>(hsi16, hsinorm, top3v, top3i);
  k_scatter<<<NN_ / 256, 256, 0, stream>>>(top3v, top3i, hsif, colsum, hid2f);
  k_h2fix<<<NN_ / 256, 256, 0, stream>>>(hid2f, colsum, hsinorm, hsif, h216,
                                         h2T16, h2n, valid2);
  // ---- flash softmax attention over hidden2 ----
  k_flash<<<NN_ / 16, 256, 0, stream>>>(hsi16, hsinorm, h216, h2T16, h2n,
                                        valid2, nullptr, hspre16);
  // ---- hs chain ----
  k_gemm_nt<<<dim3(NN_ / 128, 1), 256, 0, stream>>>(hspre16, HH_, hsw16, HH_,
                                                    hs_b, hshf, hsh16, nullptr,
                                                    NN_, HH_, HH_, 0);
  k_gemm_nt<<<dim3(NN_ / 128, 1), 256, 0, stream>>>(hsh16, HH_, hsbw16, HH_,
                                                    hs_back_b, hback, nullptr,
                                                    nullptr, NN_, HH_, HH_, 0);
  k_gemm_nt<<<dim3(NN_ / 128, 1), 256, 0, stream>>>(hsh16, HH_, hsfw16, HH_,
                                                    hs_fore_b, ouths, nullptr,
                                                    nullptr, NN_, HH_, HH_, 1);
  // ---- indi chain + final ----
  k_indi<<<(NN_ * HH_ + 255) / 256, 256, 0, stream>>>(xh, pback, hback, indi16,
                                                      NN_ * HH_);
  k_gemm_nt<<<dim3(NN_ / 128, 1), 256, 0, stream>>>(indi16, HH_, indiw16, HH_,
                                                    indi_b, outin, nullptr,
                                                    nullptr, NN_, HH_, HH_, 1);
  k_final<<<NN_ / 256, 256, 0, stream>>>(outps, ouths, outin, out_w, out_b,
                                         (float*)d_out);
}
